// GroupedQueryAttention_13984413516287
// MI455X (gfx1250) — matrix operand order repaired and checked by reading, not compiled, not run on hardware
//
#include <hip/hip_runtime.h>

// ---------------------------------------------------------------------------
// MI455X (gfx1250) GQA forward: bf16 WMMA + async global->LDS tile copies.
// ---------------------------------------------------------------------------

typedef __bf16 bf16;
typedef __attribute__((ext_vector_type(16))) __bf16 v16bf;
typedef __attribute__((ext_vector_type(8)))  __bf16 v8bf;
typedef __attribute__((ext_vector_type(8)))  float  v8f;

static constexpr int kB   = 2;
static constexpr int kT   = 2048;
static constexpr int kBT  = kB * kT;      // 4096 rows
static constexpr int kC   = 2048;         // d_model
static constexpr int kH   = 16;           // query heads
static constexpr int kKV  = 2;            // kv heads
static constexpr int kD   = 128;          // head dim
static constexpr int kKVC = kKV * kD;     // 256

static __device__ __forceinline__ bf16 f2bf(float f) {
  unsigned u = __builtin_bit_cast(unsigned, f);
  u += 0x7FFFu + ((u >> 16) & 1u);                  // round-to-nearest-even
  unsigned short s = (unsigned short)(u >> 16);
  return __builtin_bit_cast(bf16, s);
}

static __device__ __forceinline__ v16bf make16(v8bf lo, v8bf hi) {
  v16bf r;
#pragma unroll
  for (int i = 0; i < 8; ++i) { r[i] = lo[i]; r[i + 8] = hi[i]; }
  return r;
}

static __device__ __forceinline__ v8f wmma_bf16(v16bf a, v16bf b, v8f c) {
  return __builtin_amdgcn_wmma_f32_16x16x32_bf16(
      /*neg_a=*/false, a, /*neg_b=*/false, b,
      /*c_mod=*/(short)0, c, /*reuse_a=*/false, /*reuse_b=*/false);
}

// Async global->LDS 16B copy (per lane). VDST carries the LDS byte address
// (low 32 bits of the generic pointer to an LDS object); tracked by ASYNCcnt.
static __device__ __forceinline__ void async_copy_b128(const void* gptr, void* lptr) {
  unsigned lds = (unsigned)(size_t)lptr;
  asm volatile("global_load_async_to_lds_b128 %0, %1, off"
               :: "v"(lds), "v"(gptr) : "memory");
}

static __device__ __forceinline__ void wait_asynccnt0() {
  asm volatile("s_wait_asynccnt 0x0" ::: "memory");
}

// ---------------------------------------------------------------------------
// fp32 -> bf16 conversion (bandwidth-bound, trivial)
// ---------------------------------------------------------------------------
__global__ __launch_bounds__(256) void cvt_f32_bf16(const float* __restrict__ in,
                                                    bf16* __restrict__ out, int n) {
  int i = blockIdx.x * 256 + threadIdx.x;
  if (i < n) out[i] = f2bf(in[i]);
}

// ---------------------------------------------------------------------------
// bf16 GEMM: out[M,N] = A[M,K] @ Bw[K,N] + bias[N]
// Block tile 128x64, BK=32. 256 threads = 8 waves in a 4(m) x 2(n) grid,
// each wave owns a 32x32 sub-tile = 2x2 WMMA fragments.
// A tile copied with async global->LDS; B tile transposed through VGPRs.
// ---------------------------------------------------------------------------
template <bool OUT_BF16>
__global__ __launch_bounds__(256)
void gemm_bf16_kernel(const bf16* __restrict__ A, const bf16* __restrict__ Bw,
                      const float* __restrict__ bias, void* __restrict__ outp,
                      int M, int N, int K) {
  alignas(16) __shared__ bf16 As[128][40];   // row-major A tile, padded
  alignas(16) __shared__ bf16 Bs[64][40];    // column-major B tile (Bs[n][k])

  const int t = threadIdx.x;
  const int lane = t & 31, wave = t >> 5;
  const int wm = wave >> 1, wn = wave & 1;
  const int lm = lane & 15, half = lane >> 4;
  const int m0 = blockIdx.y * 128;
  const int n0 = blockIdx.x * 64;

  v8f acc[2][2] = {};

  for (int k0 = 0; k0 < K; k0 += 32) {
    // A tile: 128 rows x 32 cols -- async straight copy, 2x16B per thread
#pragma unroll
    for (int i = 0; i < 2; ++i) {
      int idx = t + i * 256;              // 0..511
      int r = idx >> 2;                   // 4 chunks of 8 bf16 per row
      int c = (idx & 3) * 8;
      async_copy_b128(&A[(size_t)(m0 + r) * K + k0 + c], &As[r][c]);
    }
    // B tile: 32 x 64, transposed into Bs[n][k] (VGPR path; overlaps async)
    {
      int r = t >> 3;                     // k row 0..31
      int c = (t & 7) * 8;                // n chunk
      v8bf v = *(const v8bf*)&Bw[(size_t)(k0 + r) * N + n0 + c];
#pragma unroll
      for (int e = 0; e < 8; ++e) Bs[c + e][r] = v[e];
    }
    wait_asynccnt0();
    __syncthreads();

    v16bf af[2], bfg[2];
#pragma unroll
    for (int i = 0; i < 2; ++i) {
      int row = wm * 32 + i * 16 + lm;
      af[i] = make16(*(const v8bf*)&As[row][half * 8],
                     *(const v8bf*)&As[row][16 + half * 8]);
    }
#pragma unroll
    for (int j = 0; j < 2; ++j) {
      int col = wn * 32 + j * 16 + lm;
      bfg[j] = make16(*(const v8bf*)&Bs[col][8 * half],
                      *(const v8bf*)&Bs[col][16 + 8 * half]);
    }
#pragma unroll
    for (int i = 0; i < 2; ++i)
#pragma unroll
      for (int j = 0; j < 2; ++j)
        acc[i][j] = wmma_bf16(af[i], bfg[j], acc[i][j]);
    __syncthreads();
  }

  // epilogue: C layout = lane n = lane%16, reg r holds row r + 8*(lane/16)
#pragma unroll
  for (int i = 0; i < 2; ++i) {
#pragma unroll
    for (int j = 0; j < 2; ++j) {
      int col = n0 + wn * 32 + j * 16 + lm;
      float bv = bias[col];
#pragma unroll
      for (int r = 0; r < 8; ++r) {
        int row = m0 + wm * 32 + i * 16 + r + 8 * half;
        float v = acc[i][j][r] + bv;
        if (OUT_BF16)
          ((bf16*)outp)[(size_t)row * N + col] = f2bf(v);
        else
          ((float*)outp)[(size_t)row * N + col] = v;
      }
    }
  }
}

// ---------------------------------------------------------------------------
// Flash-style causal GQA attention.
// Grid: (T/64, N_HEADS, B); block 128 = 4 waves; wave w owns q rows
// [q0+16w, q0+16w+16). Streams 64-key tiles with online softmax.
// K tile copied with async global->LDS; V tile transposed through VGPRs.
// ---------------------------------------------------------------------------
__global__ __launch_bounds__(128)
void gqa_attn_kernel(const bf16* __restrict__ Q, const bf16* __restrict__ Kx,
                     const bf16* __restrict__ Vx, bf16* __restrict__ Out) {
  const int qt = blockIdx.x;
  const int h  = blockIdx.y;
  const int b  = blockIdx.z;
  const int kv = h >> 3;                 // N_GROUPS = 8, head h = kv*8 + g

  const int t = threadIdx.x, lane = t & 31, w = t >> 5;
  const int lm = lane & 15, half = lane >> 4;
  const int q0 = qt * 64;
  const int qs = q0 + w * 16;            // this wave's q-strip base

  alignas(16) __shared__ bf16 Ks[64][136];     // K tile rows (== B col-major for Q.K^T)
  alignas(16) __shared__ bf16 Vt[128][72];     // V tile transposed: Vt[d][s]
  alignas(16) __shared__ bf16 Pw[4][16][72];   // per-wave P re-striping buffer

  // Q fragments for this wave's 16 rows: 4 k-steps over d=128
  const bf16* qrow = Q + ((size_t)(b * kT + qs + lm) * kC) + (size_t)h * kD;
  v16bf qf[4];
#pragma unroll
  for (int s = 0; s < 4; ++s)
    qf[s] = make16(*(const v8bf*)(qrow + s * 32 + half * 8),
                   *(const v8bf*)(qrow + s * 32 + 16 + half * 8));

  float mrow[8], lrow[8];
  v8f o[8] = {};
#pragma unroll
  for (int r = 0; r < 8; ++r) { mrow[r] = -INFINITY; lrow[r] = 0.f; }

  const float scale = 0.08838834764831845f;    // 1/sqrt(128)
  const int nTiles = qt + 1;                   // causal: keys up to q0+63

  for (int it = 0; it < nTiles; ++it) {
    const int s0 = it * 64;
    // K tile: 64 rows x 128 bf16 -- async straight copy, 8x16B per thread
#pragma unroll
    for (int i = 0; i < 8; ++i) {
      int idx = t + i * 128;
      int r = idx >> 4;
      int c = (idx & 15) * 8;
      async_copy_b128(&Kx[((size_t)(b * kT + s0 + r) * kKVC) + kv * kD + c],
                      &Ks[r][c]);
    }
    // V tile, transposed into Vt[d][s] (VGPR path; overlaps the async K copy)
#pragma unroll
    for (int i = 0; i < 8; ++i) {
      int idx = t + i * 128;
      int r = idx >> 4;
      int c = (idx & 15) * 8;
      v8bf v = *(const v8bf*)&Vx[((size_t)(b * kT + s0 + r) * kKVC) + kv * kD + c];
#pragma unroll
      for (int e = 0; e < 8; ++e) Vt[c + e][r] = v[e];
    }
    wait_asynccnt0();
    __syncthreads();

    // S = Q K^T  (16 x 64 per wave, 4 n-frags x 4 k-steps)
    v8f sacc[4] = {};
#pragma unroll
    for (int j = 0; j < 4; ++j) {
#pragma unroll
      for (int d = 0; d < 4; ++d) {
        v16bf kf = make16(*(const v8bf*)&Ks[j * 16 + lm][d * 32 + 8 * half],
                          *(const v8bf*)&Ks[j * 16 + lm][d * 32 + 16 + 8 * half]);
        sacc[j] = wmma_bf16(qf[d], kf, sacc[j]);
      }
    }

    // scale + causal mask
    const bool needMask = (s0 + 63 > qs);
    float pv[4][8];
#pragma unroll
    for (int j = 0; j < 4; ++j) {
      int key = s0 + j * 16 + lm;
#pragma unroll
      for (int r = 0; r < 8; ++r) {
        float x = sacc[j][r] * scale;
        if (needMask && key > qs + r + 8 * half) x = -INFINITY;
        pv[j][r] = x;
      }
    }

    // online softmax update (row stats per (reg r, half))
#pragma unroll
    for (int r = 0; r < 8; ++r) {
      float mx = fmaxf(fmaxf(pv[0][r], pv[1][r]), fmaxf(pv[2][r], pv[3][r]));
#pragma unroll
      for (int d = 1; d < 16; d <<= 1) mx = fmaxf(mx, __shfl_xor(mx, d, 32));
      float nm = fmaxf(mrow[r], mx);
      float alpha = __expf(mrow[r] - nm);
      float rs = 0.f;
#pragma unroll
      for (int j = 0; j < 4; ++j) {
        float p = __expf(pv[j][r] - nm);
        pv[j][r] = p;
        rs += p;
      }
#pragma unroll
      for (int d = 1; d < 16; d <<= 1) rs += __shfl_xor(rs, d, 32);
      lrow[r] = lrow[r] * alpha + rs;
      mrow[r] = nm;
#pragma unroll
      for (int j = 0; j < 8; ++j) o[j][r] *= alpha;
    }

    // re-stripe P (C layout) -> A layout through wave-private LDS
#pragma unroll
    for (int j = 0; j < 4; ++j)
#pragma unroll
      for (int r = 0; r < 8; ++r)
        Pw[w][r + 8 * half][j * 16 + lm] = f2bf(pv[j][r]);

    v16bf pf[2];
#pragma unroll
    for (int ks = 0; ks < 2; ++ks)
      pf[ks] = make16(*(const v8bf*)&Pw[w][lm][ks * 32 + half * 8],
                      *(const v8bf*)&Pw[w][lm][ks * 32 + 16 + half * 8]);

    // O += P V  (16 x 128 per wave, 8 n-frags x 2 k-steps)
#pragma unroll
    for (int j = 0; j < 8; ++j) {
#pragma unroll
      for (int ks = 0; ks < 2; ++ks) {
        v16bf vf = make16(*(const v8bf*)&Vt[j * 16 + lm][ks * 32 + 8 * half],
                          *(const v8bf*)&Vt[j * 16 + lm][ks * 32 + 16 + 8 * half]);
        o[j] = wmma_bf16(pf[ks], vf, o[j]);
      }
    }
    __syncthreads();
  }

  // normalize and store bf16 head output, layout [b*T + q][h*128 + d]
  float inv[8];
#pragma unroll
  for (int r = 0; r < 8; ++r) inv[r] = 1.0f / lrow[r];
#pragma unroll
  for (int j = 0; j < 8; ++j)
#pragma unroll
    for (int r = 0; r < 8; ++r) {
      size_t row = (size_t)(b * kT + qs + r + 8 * half);
      Out[row * kC + (size_t)h * kD + j * 16 + lm] = f2bf(o[j][r] * inv[r]);
    }
}

// ---------------------------------------------------------------------------
// Host orchestration
// ---------------------------------------------------------------------------
extern "C" void kernel_launch(void* const* d_in, const int* in_sizes, int n_in,
                              void* d_out, int out_size, void* d_ws, size_t ws_size,
                              hipStream_t stream) {
  (void)in_sizes; (void)n_in; (void)out_size; (void)ws_size;
  const float* x  = (const float*)d_in[0];
  const float* Wq = (const float*)d_in[1];
  const float* bq = (const float*)d_in[2];
  const float* Wk = (const float*)d_in[3];
  const float* bk = (const float*)d_in[4];
  const float* Wv = (const float*)d_in[5];
  const float* bv = (const float*)d_in[6];
  const float* Wo = (const float*)d_in[7];
  const float* bo = (const float*)d_in[8];

  char* ws = (char*)d_ws;
  size_t off = 0;
  auto carve = [&](size_t bytes) -> char* {
    char* p = ws + off;
    off += (bytes + 255) & ~(size_t)255;
    return p;
  };
  bf16* xb  = (bf16*)carve((size_t)kBT * kC * 2);
  bf16* Wqb = (bf16*)carve((size_t)kC * kC * 2);
  bf16* Wkb = (bf16*)carve((size_t)kC * kKVC * 2);
  bf16* Wvb = (bf16*)carve((size_t)kC * kKVC * 2);
  bf16* Wob = (bf16*)carve((size_t)kC * kC * 2);
  bf16* Qb  = (bf16*)carve((size_t)kBT * kC * 2);
  bf16* Kb  = (bf16*)carve((size_t)kBT * kKVC * 2);
  bf16* Vb  = (bf16*)carve((size_t)kBT * kKVC * 2);
  bf16* Ab  = (bf16*)carve((size_t)kBT * kC * 2);

  auto cvt = [&](const float* in, bf16* out, int n) {
    cvt_f32_bf16<<<(n + 255) / 256, 256, 0, stream>>>(in, out, n);
  };
  cvt(x,  xb,  kBT * kC);
  cvt(Wq, Wqb, kC * kC);
  cvt(Wk, Wkb, kC * kKVC);
  cvt(Wv, Wvb, kC * kKVC);
  cvt(Wo, Wob, kC * kC);

  // QKV projections
  gemm_bf16_kernel<true><<<dim3(kC / 64,   kBT / 128), 256, 0, stream>>>(
      xb, Wqb, bq, (void*)Qb, kBT, kC,   kC);
  gemm_bf16_kernel<true><<<dim3(kKVC / 64, kBT / 128), 256, 0, stream>>>(
      xb, Wkb, bk, (void*)Kb, kBT, kKVC, kC);
  gemm_bf16_kernel<true><<<dim3(kKVC / 64, kBT / 128), 256, 0, stream>>>(
      xb, Wvb, bv, (void*)Vb, kBT, kKVC, kC);

  // causal grouped-query attention
  gqa_attn_kernel<<<dim3(kT / 64, kH, kB), 128, 0, stream>>>(Qb, Kb, Vb, Ab);

  // output projection -> fp32 d_out
  gemm_bf16_kernel<false><<<dim3(kC / 64, kBT / 128), 256, 0, stream>>>(
      Ab, Wob, bo, d_out, kBT, kC, kC);
}
